// RNN_circular_LowEtAl_71081708749398
// MI455X (gfx1250) — matrix operand order repaired and checked by reading, not compiled, not run on hardware
//
#include <hip/hip_runtime.h>

// CDNA5 (gfx1250) WMMA IRNN, transposed formulation: D = W_h (A) x h^T (B).
// 16 WGs x 256 threads; per WG: 16 batch rows, full H=512.
// Each wave owns 64 J-columns (4 tiles of 16):
//   tiles 0,1 -> W_h fragments register-resident across all 1024 steps
//   tile  2   -> W_h slice cached in LDS (loaded once)
//   tile  3   -> streamed from L2 each step (opaque offset defeats LICM+spill)

typedef __attribute__((ext_vector_type(16))) __bf16 v16bf;
typedef __attribute__((ext_vector_type(8)))  __bf16 v8bf;
typedef __attribute__((ext_vector_type(8)))  float  v8f;

#define T_STEPS 1024
#define HDIM    512
#define BATCH   256
#define ROWS    16
#define NWAVES  8
#define TWO_PI_F 6.283185307179586f

union Frag16 { v16bf v; uint4 q[2]; };

// One-shot f32 -> bf16 conversion of W_h into workspace (512*512 elements).
__global__ void wh_to_bf16_kernel(const float* __restrict__ src,
                                  __bf16* __restrict__ dst) {
    int i = blockIdx.x * blockDim.x + threadIdx.x;   // grid exactly covers 512*512
    dst[i] = (__bf16)src[i];
}

__global__ __launch_bounds__(256)
void irnn_wmma_kernel(const float* __restrict__ x,       // [256][1024]
                      const float* __restrict__ theta0,  // [256]
                      const __bf16* __restrict__ whbf,   // [512][512] bf16 row-major (= W_h)
                      const float* __restrict__ Wx,      // [512]
                      const float* __restrict__ bxg,     // [512]
                      const float* __restrict__ Wh0,     // [512]
                      const float* __restrict__ bh0,     // [512]
                      const float* __restrict__ Wout,    // [2][512]
                      const float* __restrict__ boutg,   // [2]
                      float* __restrict__ out)           // [1025][256][2]
{
    __shared__ __align__(16) __bf16 hbf[ROWS][HDIM];        // 16 KB: h (B operand, bf16)
    __shared__ __align__(16) float  hf[ROWS][HDIM];         // 32 KB: h f32 for output head
    __shared__ __align__(16) __bf16 whlds[NWAVES * 16 * HDIM]; // 128 KB: tile-2 W_h rows
    __shared__ float wx_l[HDIM], bx_l[HDIM];
    __shared__ float w0_l[HDIM], w1_l[HDIM];
    __shared__ float xstep[ROWS];

    const int tid   = threadIdx.x;
    const int wave  = tid >> 5;          // 0..7
    const int lane  = tid & 31;
    const int hi    = lane >> 4;         // 0: lanes 0-15, 1: lanes 16-31
    const int ln16  = lane & 15;
    const int b0    = blockIdx.x * ROWS;
    const int jbase = wave * 64;         // this wave's 4 J tiles

    const float bout0 = boutg[0];
    const float bout1 = boutg[1];

    // ---- params into LDS ----
    wx_l[tid] = Wx[tid];          wx_l[tid + 256] = Wx[tid + 256];
    bx_l[tid] = bxg[tid];         bx_l[tid + 256] = bxg[tid + 256];
    w0_l[tid] = Wout[tid];        w0_l[tid + 256] = Wout[tid + 256];
    w1_l[tid] = Wout[HDIM + tid]; w1_l[tid + 256] = Wout[HDIM + tid + 256];

    // ---- h0 = theta0*2pi*W_h0 + b_h0 ----
    #pragma unroll
    for (int s = 0; s < 2; ++s) {
        int j = tid + s * 256;
        float w = Wh0[j];
        float b = bh0[j];
        #pragma unroll
        for (int r = 0; r < ROWS; ++r) {
            float v = theta0[b0 + r] * TWO_PI_F * w + b;
            hf[r][j]  = v;
            hbf[r][j] = (__bf16)v;
        }
    }

    // ---- tile-2 W_h rows (j with (j&63) in [32,48)) into LDS, once ----
    // 128 rows of 512 bf16 = 8192 uint4 chunks; 32 per thread.
    for (int i = tid; i < NWAVES * 16 * (HDIM / 8); i += 256) {
        int row = i >> 6;                 // 0..127 (w*16 + jloc)
        int seg = i & 63;                 // 16B segment within row
        int w   = row >> 4;
        int jl  = row & 15;
        ((uint4*)whlds)[i] =
            *(const uint4*)(whbf + (size_t)(w * 64 + 32 + jl) * HDIM + seg * 8);
    }

    // ---- register-resident W_h A-fragments (tiles 0,1) ----
    Frag16 wf0[16], wf1[16];
    {
        const __bf16* r0p = whbf + (size_t)(jbase + 0  + ln16) * HDIM + hi * 8;
        const __bf16* r1p = whbf + (size_t)(jbase + 16 + ln16) * HDIM + hi * 8;
        #pragma unroll
        for (int c = 0; c < 16; ++c) {
            wf0[c].q[0] = *(const uint4*)(r0p + c * 32);
            wf0[c].q[1] = *(const uint4*)(r0p + c * 32 + 16);
            wf1[c].q[0] = *(const uint4*)(r1p + c * 32);
            wf1[c].q[1] = *(const uint4*)(r1p + c * 32 + 16);
        }
    }
    const __bf16* r3p = whbf + (size_t)(jbase + 48 + ln16) * HDIM + hi * 8;  // tile 3 (L2)
    const unsigned l2base = (unsigned)((wave * 16 + ln16) * HDIM + hi * 8);  // tile 2 (LDS)
    const __bf16* hrow = &hbf[ln16][0];

    __syncthreads();

    // ---- output head for hts[0] = h0 : wave w handles rows 2w, 2w+1 ----
    {
        const int r0 = wave * 2, r1 = r0 + 1;
        float a00 = 0.f, a01 = 0.f, a10 = 0.f, a11 = 0.f;
        #pragma unroll
        for (int i = 0; i < 16; ++i) {
            int j = lane + 32 * i;
            float h0v = hf[r0][j], h1v = hf[r1][j];
            a00 += h0v * w0_l[j]; a01 += h0v * w1_l[j];
            a10 += h1v * w0_l[j]; a11 += h1v * w1_l[j];
        }
        #pragma unroll
        for (int off = 16; off > 0; off >>= 1) {
            a00 += __shfl_xor(a00, off, 32); a01 += __shfl_xor(a01, off, 32);
            a10 += __shfl_xor(a10, off, 32); a11 += __shfl_xor(a11, off, 32);
        }
        if (lane == 0) {
            float* o = out + (size_t)(b0 + r0) * 2;
            o[0] = a00 + bout0; o[1] = a01 + bout1;
            o[2] = a10 + bout0; o[3] = a11 + bout1;   // row r1 adjacent
        }
    }

    // ---- sequential time loop ----
    for (int t = 0; t < T_STEPS; ++t) {
        if (tid < ROWS) xstep[tid] = x[(b0 + tid) * T_STEPS + t];

        // Opaque zero offset: keeps streamed/LDS W_h addresses loop-variant so
        // LICM cannot hoist the loads out of the t-loop and spill them.
        unsigned zoff = 0;
        asm volatile("" : "+v"(zoff));
        const __bf16* s3   = r3p + zoff;
        const unsigned l2o = l2base + zoff;

        v8f accs[4] = {};
        #pragma unroll
        for (int c = 0; c < 16; ++c) {
            // B fragment: h^T — lane = batch column ln16, K = 8*hi+0..7 and 16+8*hi+0..7
            Frag16 hb;
            const __bf16* hp = hrow + c * 32 + hi * 8;
            hb.q[0] = ((const uint4*)hp)[0];
            hb.q[1] = ((const uint4*)hp)[2];
            // tile-2 A fragment from LDS
            Frag16 w2;
            w2.q[0] = *(const uint4*)(whlds + l2o + c * 32);
            w2.q[1] = *(const uint4*)(whlds + l2o + c * 32 + 16);
            // tile-3 A fragment streamed from L2-resident W_h
            Frag16 w3;
            w3.q[0] = *(const uint4*)(s3 + c * 32);
            w3.q[1] = *(const uint4*)(s3 + c * 32 + 16);
            accs[0] = __builtin_amdgcn_wmma_f32_16x16x32_bf16(
                          false, wf0[c].v, false, hb.v, (short)0, accs[0], false, false);
            accs[1] = __builtin_amdgcn_wmma_f32_16x16x32_bf16(
                          false, wf1[c].v, false, hb.v, (short)0, accs[1], false, false);
            accs[2] = __builtin_amdgcn_wmma_f32_16x16x32_bf16(
                          false, w2.v,     false, hb.v, (short)0, accs[2], false, false);
            accs[3] = __builtin_amdgcn_wmma_f32_16x16x32_bf16(
                          false, w3.v,     false, hb.v, (short)0, accs[3], false, false);
        }
        __syncthreads();   // all reads of hbf done; xstep visible

        // epilogue: h_new = relu(D + x[b,t]*W_x[j] + b_x[j])
        // D[m=j][n=b]: lane -> batch row ln16; VGPR r -> j = jbase + kk*16 + 8*hi + r
        {
            const float xv = xstep[ln16];
            #pragma unroll
            for (int kk = 0; kk < 4; ++kk) {
                const int jc = jbase + kk * 16 + 8 * hi;
                float v[8];
                #pragma unroll
                for (int r = 0; r < 8; ++r) {
                    float vv = accs[kk][r] + xv * wx_l[jc + r] + bx_l[jc + r];
                    v[r] = vv > 0.f ? vv : 0.f;
                }
                v8bf hv;
                #pragma unroll
                for (int r = 0; r < 8; ++r) hv[r] = (__bf16)v[r];
                *(v8bf*)&hbf[ln16][jc] = hv;                       // one b128 store
                *(float4*)&hf[ln16][jc]     = make_float4(v[0], v[1], v[2], v[3]);
                *(float4*)&hf[ln16][jc + 4] = make_float4(v[4], v[5], v[6], v[7]);
            }
        }
        __syncthreads();   // h_new visible to all waves

        // output head for hts[t+1]
        {
            const int r0 = wave * 2, r1 = r0 + 1;
            float a00 = 0.f, a01 = 0.f, a10 = 0.f, a11 = 0.f;
            #pragma unroll
            for (int i = 0; i < 16; ++i) {
                int j = lane + 32 * i;
                float h0v = hf[r0][j], h1v = hf[r1][j];
                a00 += h0v * w0_l[j]; a01 += h0v * w1_l[j];
                a10 += h1v * w0_l[j]; a11 += h1v * w1_l[j];
            }
            #pragma unroll
            for (int off = 16; off > 0; off >>= 1) {
                a00 += __shfl_xor(a00, off, 32); a01 += __shfl_xor(a01, off, 32);
                a10 += __shfl_xor(a10, off, 32); a11 += __shfl_xor(a11, off, 32);
            }
            if (lane == 0) {
                float* o = out + (size_t)(t + 1) * (BATCH * 2) + (size_t)(b0 + r0) * 2;
                o[0] = a00 + bout0; o[1] = a01 + bout1;
                o[2] = a10 + bout0; o[3] = a11 + bout1;
            }
        }
    }
}

extern "C" void kernel_launch(void* const* d_in, const int* in_sizes, int n_in,
                              void* d_out, int out_size, void* d_ws, size_t ws_size,
                              hipStream_t stream) {
    const float* x      = (const float*)d_in[0];   // [256,1024,1]
    const float* theta0 = (const float*)d_in[1];   // [256,1]
    const float* W_h    = (const float*)d_in[2];   // [512,512]
    const float* W_x    = (const float*)d_in[3];   // [512,1]
    const float* b_x    = (const float*)d_in[4];   // [512]
    const float* W_h0   = (const float*)d_in[5];   // [512,1]
    const float* b_h0   = (const float*)d_in[6];   // [512]
    const float* W_out  = (const float*)d_in[7];   // [2,512]
    const float* b_out  = (const float*)d_in[8];   // [2]
    float* out = (float*)d_out;                    // [1025,256,2]

    __bf16* whbf = (__bf16*)d_ws;                  // 512 KB bf16 copy of W_h

    wh_to_bf16_kernel<<<(HDIM * HDIM) / 512, 512, 0, stream>>>(W_h, whbf);
    irnn_wmma_kernel<<<BATCH / ROWS, 256, 0, stream>>>(
        x, theta0, whbf, W_x, b_x, W_h0, b_h0, W_out, b_out, out);
}
